// Self_Attention_83820581749072
// MI455X (gfx1250) — matrix operand order repaired and checked by reading, not compiled, not run on hardware
//
#include <hip/hip_runtime.h>

// Problem constants (reference: B,C,H,W = 4,64,64,64)
#define Bn 4
#define Cn 64
#define Hn 64
#define Wn 64
#define Nn (Hn * Wn)   // 4096 tokens per batch

typedef __attribute__((ext_vector_type(16))) __bf16        v16bf;
typedef __attribute__((ext_vector_type(8)))  float         v8f;
typedef __attribute__((ext_vector_type(4)))  float         fvec4;
typedef __attribute__((ext_vector_type(4)))  unsigned int  ui4;
typedef __attribute__((ext_vector_type(8)))  int           i8v;
typedef __attribute__((ext_vector_type(4)))  int           i4v;

union AB32 { v16bf v; fvec4 f[2]; };

__device__ __forceinline__ fvec4 ld4h(const __bf16* p) {
    return *reinterpret_cast<const fvec4*>(p);   // 8 bf16 = 16 bytes
}
__device__ __forceinline__ v16bf pack2(fvec4 a, fvec4 b) {
    AB32 t; t.f[0] = a; t.f[1] = b; return t.v;
}
__device__ __forceinline__ v8f wmma_bf16(v16bf a, v16bf b, v8f c) {
    // D(f32 16x16) = A(bf16 16x32) * B(bf16 32x16) + C
    return __builtin_amdgcn_wmma_f32_16x16x32_bf16(false, a, false, b, (short)0, c,
                                                   false, false);
}

// ---------------------------------------------------------------------------
// Tensor Data Mover: 2-D tile load, global -> LDS.  Descriptor per ISA §8.3/8.4.
//   dim0 = contiguous element axis, dim1 = strided axis (stride0 elements).
//   data_size = 2 bytes (bf16).  Result is packed rows of tile_d0 elements.
// ---------------------------------------------------------------------------
__device__ __forceinline__ void tdm_load_2d_bf16(
    unsigned lds_addr, const __bf16* gptr,
    unsigned tensor_d0, unsigned tensor_d1, unsigned stride0,
    unsigned tile_d0, unsigned tile_d1)
{
    const unsigned long long ga = (unsigned long long)(size_t)gptr;
    ui4 g0;
    g0.x = 1u;                                      // count=1 (valid user D#)
    g0.y = lds_addr;                                // lds_addr [63:32]
    g0.z = (unsigned)(ga & 0xFFFFFFFFu);            // global_addr [95:64]
    g0.w = (unsigned)((ga >> 32) & 0x01FFFFFFu)     // global_addr [120:96]
         | 0x80000000u;                             // type=2 ("image") [127:126]
    i8v g1;
    g1[0] = (int)(1u << 16);                        // data_size=1 -> 2 bytes
    g1[1] = (int)(tensor_d0 << 16);                 // tensor_dim0[15:0]
    g1[2] = (int)((tensor_d0 >> 16) | (tensor_d1 << 16));
    g1[3] = (int)((tensor_d1 >> 16) | (tile_d0 << 16));
    g1[4] = (int)tile_d1;                           // tile_dim1 | tile_dim2=0
    g1[5] = (int)stride0;                           // tensor_dim0_stride[31:0]
    g1[6] = 0;                                      // stride0_hi | stride1_lo
    g1[7] = 0;
    const i4v z4 = {0, 0, 0, 0};
#if __clang_major__ >= 23
    const i8v z8 = {0, 0, 0, 0, 0, 0, 0, 0};
    __builtin_amdgcn_tensor_load_to_lds(g0, g1, z4, z4, z8, 0);
#else
    __builtin_amdgcn_tensor_load_to_lds(g0, g1, z4, z4, 0);
#endif
}

// ---------------------------------------------------------------------------
// Kernel 1: fused QKV 1x1-conv projection + positional fold, fp32 -> bf16.
//   Q  [B][N][C]  (query rows,  WMMA-A friendly)
//   Kt [B][N][C]  (k^T rows with rel_h+rel_w folded in, WMMA-B friendly)
//   Vb [B][C][N]  (v rows over m, WMMA-A friendly for the PV GEMM)
// ---------------------------------------------------------------------------
__global__ __launch_bounds__(256) void qkv_proj_kernel(
    const float* __restrict__ x,
    const float* __restrict__ Wq, const float* __restrict__ bq,
    const float* __restrict__ Wk, const float* __restrict__ bk,
    const float* __restrict__ Wv, const float* __restrict__ bv,
    const float* __restrict__ rel_h, const float* __restrict__ rel_w,
    __bf16* __restrict__ Q, __bf16* __restrict__ Kt, __bf16* __restrict__ Vb)
{
    __shared__ __align__(16) float xs[Cn][64];   // 16 KB tile of x

    const int b  = blockIdx.x / (Nn / 64);
    const int n0 = (blockIdx.x % (Nn / 64)) * 64;
    const int tid = threadIdx.x;

    const float* xb = x + (size_t)b * Cn * Nn;
    for (int i = tid; i < Cn * 64; i += 256) {
        const int c = i >> 6, nl = i & 63;
        xs[c][nl] = xb[(size_t)c * Nn + n0 + nl];
    }
    __syncthreads();

    const int nl = tid & 63;          // token within tile (coalesced dim)
    const int og = tid >> 6;          // output-channel group 0..3 (wave-uniform)
    const int n  = n0 + nl;
    const int h  = n >> 6;
    const int w  = n & 63;

    for (int oi = 0; oi < 16; ++oi) {
        const int o = og * 16 + oi;
        float aq = bq[o], ak = bk[o], av = bv[o];
        const float* wqr = Wq + o * Cn;
        const float* wkr = Wk + o * Cn;
        const float* wvr = Wv + o * Cn;
        #pragma unroll 8
        for (int c = 0; c < Cn; ++c) {
            const float xv = xs[c][nl];
            aq = fmaf(wqr[c], xv, aq);
            ak = fmaf(wkr[c], xv, ak);
            av = fmaf(wvr[c], xv, av);
        }
        ak += rel_h[o * Hn + h] + rel_w[o * Wn + w];   // fold positional bias into K
        Q [((size_t)b * Nn + n) * Cn + o] = (__bf16)aq;
        Kt[((size_t)b * Nn + n) * Cn + o] = (__bf16)ak;
        Vb[((size_t)b * Cn + o) * Nn + n] = (__bf16)av;
    }
}

// ---------------------------------------------------------------------------
// Kernel 2: fused flash attention.  Grid = B * (N/64) workgroups, 4 waves.
// Wave w owns query rows [n0+16w, n0+16w+16).  Per 64-key slab:
//   TDM double-buffers K/V tiles into LDS (wave0 issues, TENSORcnt tracked),
//   8 WMMAs for S = Q K^T, fp32 online softmax, 8 WMMAs for O += V P^T.
// ---------------------------------------------------------------------------
__global__ __launch_bounds__(128) void attn_fused_kernel(
    const __bf16* __restrict__ Q, const __bf16* __restrict__ Kt,
    const __bf16* __restrict__ V, float* __restrict__ out)
{
    __shared__ __align__(16) __bf16 Klds[2][64][64];   // double-buffered K tile (16 KB)
    __shared__ __align__(16) __bf16 Vlds[2][64][64];   // double-buffered V tile (16 KB)
    __shared__ __align__(16) float  Slds[4][16][64];   // per-wave S scratch   (16 KB)
    __shared__ __align__(16) __bf16 Plds[4][16][64];   // per-wave P scratch   ( 8 KB)

    const int b    = blockIdx.x >> 6;
    const int n0   = (blockIdx.x & 63) << 6;
    const int tid  = threadIdx.x;
    const int wv   = tid >> 5;        // wave 0..3
    const int lane = tid & 31;
    const int lh   = lane >> 4;       // lane half (ISA operand layouts split on this)
    const int l16  = lane & 15;

    const __bf16* Qb = Q  + (size_t)b * Nn * Cn;
    const __bf16* Kb = Kt + (size_t)b * Nn * Cn;
    const __bf16* Vb = V  + (size_t)b * Cn * Nn;

    // --- preload A(Q): 16x32 bf16 A-layout, two K=32 chunks of C=64 ---
    // lanes 0-15: elems0-7 = c 0..7,  elems8-15 = c 16..23
    // lanes16-31: elems0-7 = c 8..15, elems8-15 = c 24..31   (chunk1: +32)
    const __bf16* qrow = Qb + (size_t)(n0 + wv * 16 + l16) * Cn;
    const v16bf Aq0 = pack2(ld4h(qrow + lh * 8),      ld4h(qrow + 16 + lh * 8));
    const v16bf Aq1 = pack2(ld4h(qrow + 32 + lh * 8), ld4h(qrow + 48 + lh * 8));

    v8f O[4];
    #pragma unroll
    for (int t = 0; t < 4; ++t) O[t] = (v8f){0, 0, 0, 0, 0, 0, 0, 0};
    float m_run = -1e30f, l_run = 0.0f;

    // --- prologue: TDM tile 0 into buffer 0 ---
    if (wv == 0) {
        // K tile: rows m0..m0+64 of Kt[b] are one contiguous 8 KB block.
        tdm_load_2d_bf16((unsigned)(size_t)&Klds[0][0][0], Kb,
                         /*td0=*/Cn, /*td1=*/64, /*stride0=*/Cn, 64, 64);
        // V tile: 64 channel rows x 64 keys, row stride Nn elements.
        tdm_load_2d_bf16((unsigned)(size_t)&Vlds[0][0][0], Vb,
                         /*td0=*/64, /*td1=*/Cn, /*stride0=*/Nn, 64, 64);
        __builtin_amdgcn_s_wait_tensorcnt(0);
    }
    __syncthreads();

    const int NT = Nn / 64;
    for (int kt = 0; kt < NT; ++kt) {
        const int m0  = kt * 64;
        const int buf = kt & 1;

        // --- TDM prefetch of the next key/value tile into the other buffer ---
        if (wv == 0 && kt + 1 < NT) {
            const int m1 = m0 + 64;
            tdm_load_2d_bf16((unsigned)(size_t)&Klds[buf ^ 1][0][0],
                             Kb + (size_t)m1 * Cn, Cn, 64, Cn, 64, 64);
            tdm_load_2d_bf16((unsigned)(size_t)&Vlds[buf ^ 1][0][0],
                             Vb + m1, 64, Cn, Nn, 64, 64);
        }

        // ---- S = Q @ K^T : hoist all B operands, then an 8-WMMA burst ----
        v16bf Bk0[4], Bk1[4];
        #pragma unroll
        for (int t = 0; t < 4; ++t) {
            // B 32x16 layout: lane = column m; lanes0-15 hold c 0..7,16..23,
            // lanes16-31 hold c 8..15,24..31 (mirrors the Q layout).  chunk1: +32.
            const __bf16* krow = &Klds[buf][t * 16 + l16][lh * 8];
            Bk0[t] = pack2(ld4h(krow),      ld4h(krow + 16));
            Bk1[t] = pack2(ld4h(krow + 32), ld4h(krow + 48));
        }
        v8f S[4];
        #pragma unroll
        for (int t = 0; t < 4; ++t) {
            v8f s = (v8f){0, 0, 0, 0, 0, 0, 0, 0};
            s = wmma_bf16(Aq0, Bk0[t], s);
            s = wmma_bf16(Aq1, Bk1[t], s);
            S[t] = s;
        }
        // C/D layout: vgpr v, lane-half lh -> row v+8*lh, col l16
        #pragma unroll
        for (int t = 0; t < 4; ++t)
            #pragma unroll
            for (int v = 0; v < 8; ++v)
                Slds[wv][v + 8 * lh][t * 16 + l16] = S[t][v];

        // ---- online softmax over this 64-key slab (fp32) ----
        // lane pair (l16, l16+16) owns query row l16; each lane scans 32 cols.
        const float* srow = &Slds[wv][l16][lh * 32];
        float mx = -1e30f;
        #pragma unroll 8
        for (int j = 0; j < 32; ++j) mx = fmaxf(mx, srow[j]);
        mx = fmaxf(mx, __shfl_xor(mx, 16, 32));
        const float m_new = fmaxf(m_run, mx);
        const float scale = __expf(m_run - m_new);
        float sum = 0.0f;
        __bf16* prow = &Plds[wv][l16][lh * 32];
        #pragma unroll 8
        for (int j = 0; j < 32; ++j) {
            const float p = __expf(srow[j] - m_new);
            sum += p;
            prow[j] = (__bf16)p;
        }
        sum += __shfl_xor(sum, 16, 32);
        l_run = l_run * scale + sum;
        m_run = m_new;

        #pragma unroll
        for (int t = 0; t < 4; ++t) O[t] *= scale;

        // ---- O += V @ P^T : contraction over the 64 keys of this slab ----
        // B(P) 32x16: lane column = query l16; lanes0-15 m 0..7,16..23, lanes16-31 m 8..15,24..31.
        const __bf16* pbase = &Plds[wv][l16][lh * 8];
        const v16bf Bp0 = pack2(ld4h(pbase),      ld4h(pbase + 16));
        const v16bf Bp1 = pack2(ld4h(pbase + 32), ld4h(pbase + 48));
        v16bf Av0[4], Av1[4];
        #pragma unroll
        for (int t = 0; t < 4; ++t) {
            // A(V) 16x32: row c = 16t + l16; m chunks mirror the Q layout.
            const __bf16* vrow = &Vlds[buf][t * 16 + l16][0];
            Av0[t] = pack2(ld4h(vrow + lh * 8),      ld4h(vrow + 16 + lh * 8));
            Av1[t] = pack2(ld4h(vrow + 32 + lh * 8), ld4h(vrow + 48 + lh * 8));
        }
        #pragma unroll
        for (int t = 0; t < 4; ++t) {
            O[t] = wmma_bf16(Av0[t], Bp0, O[t]);
            O[t] = wmma_bf16(Av1[t], Bp1, O[t]);
        }

        // --- close the slab: next tile's DMA must be done, buffers reusable ---
        if (wv == 0) __builtin_amdgcn_s_wait_tensorcnt(0);
        __syncthreads();
    }

    // ---- epilogue: normalize by row sum, store out[b][c][n] ----
    const float inv = 1.0f / l_run;           // l_run is for query col = l16
    const int   n   = n0 + wv * 16 + l16;
    #pragma unroll
    for (int t = 0; t < 4; ++t)
        #pragma unroll
        for (int v = 0; v < 8; ++v) {
            const int c = t * 16 + v + 8 * lh;
            out[((size_t)b * Cn + c) * Nn + n] = O[t][v] * inv;
        }
}

// ---------------------------------------------------------------------------
extern "C" void kernel_launch(void* const* d_in, const int* in_sizes, int n_in,
                              void* d_out, int out_size, void* d_ws, size_t ws_size,
                              hipStream_t stream) {
    const float* x     = (const float*)d_in[0];
    const float* Wq    = (const float*)d_in[1];
    const float* bq    = (const float*)d_in[2];
    const float* Wk    = (const float*)d_in[3];
    const float* bk    = (const float*)d_in[4];
    const float* Wv    = (const float*)d_in[5];
    const float* bv    = (const float*)d_in[6];
    const float* rel_h = (const float*)d_in[7];
    const float* rel_w = (const float*)d_in[8];
    float* out = (float*)d_out;

    // workspace: Q (2 MB) | Kt (2 MB) | Vb (2 MB), all bf16
    __bf16* Qw  = (__bf16*)d_ws;
    __bf16* Ktw = Qw  + (size_t)Bn * Nn * Cn;
    __bf16* Vw  = Ktw + (size_t)Bn * Nn * Cn;

    qkv_proj_kernel<<<Bn * (Nn / 64), 256, 0, stream>>>(
        x, Wq, bq, Wk, bk, Wv, bv, rel_h, rel_w, Qw, Ktw, Vw);
    attn_fused_kernel<<<Bn * (Nn / 64), 128, 0, stream>>>(Qw, Ktw, Vw, out);
}
